// ChimeraLayer_9234179686609
// MI455X (gfx1250) — compile-verified, matrix operand order checked by reading, not run on hardware
//
#include <hip/hip_runtime.h>
#include <hip/hip_bf16.h>

typedef __bf16 bf16_t;
typedef __attribute__((ext_vector_type(16))) __bf16 v16bf;
typedef __attribute__((ext_vector_type(8)))  __bf16 v8bf;
typedef __attribute__((ext_vector_type(8)))  float  v8f;

#define B_SZ      32
#define S_LEN     256
#define DM        1024
#define DI        2048
#define NH        8
#define DSTATE    64
#define NC        4
#define LCH       64
#define HDIM      256
#define PROJ_N    5128
#define PROJ_LD   5136
#define OFF_X     2048
#define OFF_B     4096
#define OFF_C     4608
#define OFF_DT    5120
#define EPS_F     1.1920929e-07f

// ---------------------------------------------------------------------------
// WMMA helpers (CDNA5 v_wmma_f32_16x16x32_bf16, wave32)
// ---------------------------------------------------------------------------
__device__ __forceinline__ v8f wmma_bf16(v16bf a, v16bf b, v8f c) {
  return __builtin_amdgcn_wmma_f32_16x16x32_bf16(
      /*neg_a=*/false, a, /*neg_b=*/false, b,
      /*c_mod=*/(short)0, c, /*reuse_a=*/false, /*reuse_b=*/false);
}

// Holder-major fragment (A rows, or B columns that are contiguous rows of a
// row-major (N,K) array).  Per ISA 7.12.2 (16-bit 16x32 operand):
//   holder index = r + lane%16
//   element e holds K = k0 + (e<8 ? e : e+8) + (lane<16 ? 0 : 8)
// Lowered to two 16B loads (global_load_b128 / ds_load_b128).
__device__ __forceinline__ v16bf frag_rm(const bf16_t* base, int ld, int r,
                                         int k0, int lane, int rmax) {
  int row = r + (lane & 15);
  row = row > rmax ? rmax : row;
  const bf16_t* p = base + (size_t)row * ld + (size_t)(k0 + ((lane >> 4) << 3));
  v8bf lo = *reinterpret_cast<const v8bf*>(p);
  v8bf hi = *reinterpret_cast<const v8bf*>(p + 16);
  v16bf f;
#pragma unroll
  for (int i = 0; i < 8; ++i) { f[i] = lo[i]; f[i + 8] = hi[i]; }
  return f;
}

__device__ __forceinline__ float silu_f(float x) {
  return x / (1.f + __expf(-x));
}

// ---------------------------------------------------------------------------
// 1) fp32 -> bf16 cast (weights)
// ---------------------------------------------------------------------------
__global__ __launch_bounds__(256) void k_cast_bf16(const float* __restrict__ in,
                                                   bf16_t* __restrict__ out,
                                                   size_t n) {
  size_t i = (size_t)blockIdx.x * 256 + threadIdx.x;
  size_t stride = (size_t)gridDim.x * 256;
  for (; i < n; i += stride) out[i] = (bf16_t)in[i];
}

// ---------------------------------------------------------------------------
// 2) RMSNorm over D_MODEL, emit bf16 activations
// ---------------------------------------------------------------------------
__global__ __launch_bounds__(256) void k_rms_in(const float* __restrict__ x,
                                                const float* __restrict__ g,
                                                bf16_t* __restrict__ out) {
  int row = blockIdx.x, t = threadIdx.x;
  __shared__ float red[256];
  float s = 0.f;
  for (int i = t; i < DM; i += 256) {
    float v = x[(size_t)row * DM + i];
    s += v * v;
  }
  red[t] = s; __syncthreads();
  for (int off = 128; off > 0; off >>= 1) {
    if (t < off) red[t] += red[t + off];
    __syncthreads();
  }
  float rs = rsqrtf(red[0] / (float)DM + EPS_F);
  for (int i = t; i < DM; i += 256)
    out[(size_t)row * DM + i] = (bf16_t)(x[(size_t)row * DM + i] * rs * g[i]);
}

// ---------------------------------------------------------------------------
// 3/10) bf16 WMMA GEMM:  C[M,N](+residual) = A[M,K] * W[N,K]^T
//   block 256 thr = 8 waves -> 128(M) x 128(N) tile; wave = 32x64 (2x4 accs)
// ---------------------------------------------------------------------------
__global__ __launch_bounds__(256) void k_gemm(const bf16_t* __restrict__ A,
                                              const bf16_t* __restrict__ W,
                                              const float* __restrict__ Rz,
                                              float* __restrict__ C,
                                              int M, int N, int K, int ldc) {
  int lane = threadIdx.x & 31;
  int wave = threadIdx.x >> 5;
  int wm = wave & 3, wn = wave >> 2;
  int m0 = blockIdx.y * 128 + wm * 32;
  int n0 = blockIdx.x * 128 + wn * 64;

  v8f zero = {0.f, 0.f, 0.f, 0.f, 0.f, 0.f, 0.f, 0.f};
  v8f acc[2][4];
#pragma unroll
  for (int i = 0; i < 2; ++i)
#pragma unroll
    for (int j = 0; j < 4; ++j) acc[i][j] = zero;

  for (int k0 = 0; k0 < K; k0 += 32) {
    v16bf a0 = frag_rm(A, K, m0,      k0, lane, M - 1);
    v16bf a1 = frag_rm(A, K, m0 + 16, k0, lane, M - 1);
#pragma unroll
    for (int j = 0; j < 4; ++j) {
      v16bf bw = frag_rm(W, K, n0 + j * 16, k0, lane, N - 1);
      acc[0][j] = wmma_bf16(a0, bw, acc[0][j]);
      acc[1][j] = wmma_bf16(a1, bw, acc[1][j]);
    }
  }

  int rlo = (lane >> 4) << 3;
#pragma unroll
  for (int tmi = 0; tmi < 2; ++tmi)
#pragma unroll
    for (int j = 0; j < 4; ++j)
#pragma unroll
      for (int r = 0; r < 8; ++r) {
        int row = m0 + tmi * 16 + r + rlo;
        int col = n0 + j * 16 + (lane & 15);
        if (row < M && col < N) {
          float v = acc[tmi][j][r];
          if (Rz) v += Rz[(size_t)row * N + col];
          C[(size_t)row * ldc + col] = v;
        }
      }
}

// ---------------------------------------------------------------------------
// 4) depthwise causal conv(4)+SiLU -> xbf ; dt proj + per-head mean + softplus
// ---------------------------------------------------------------------------
__global__ __launch_bounds__(256) void k_conv_dt(const float* __restrict__ proj,
                                                 const float* __restrict__ conv_w,
                                                 const float* __restrict__ conv_b,
                                                 const float* __restrict__ W_dt,
                                                 const float* __restrict__ b_dt,
                                                 bf16_t* __restrict__ xbf,
                                                 float* __restrict__ dt_h) {
  int row = blockIdx.x;
  int b = row / S_LEN, s = row % S_LEN;
  int t = threadIdx.x;
  __shared__ float dtr[8];
  __shared__ float red[256];
  if (t < 8) dtr[t] = proj[(size_t)row * PROJ_LD + OFF_DT + t];
  __syncthreads();

  float dtsum = 0.f;
#pragma unroll
  for (int cc = 0; cc < 8; ++cc) {
    int ch = t * 8 + cc;
    float acc = conv_b[ch];
#pragma unroll
    for (int j = 0; j < 4; ++j) {
      int sp = s + j - 3;
      if (sp >= 0)
        acc += conv_w[ch * 4 + j] *
               proj[(size_t)(b * S_LEN + sp) * PROJ_LD + OFF_X + ch];
    }
    xbf[(size_t)row * DI + ch] = (bf16_t)silu_f(acc);
    float dv = b_dt[ch];
#pragma unroll
    for (int k = 0; k < 8; ++k) dv += dtr[k] * W_dt[ch * 8 + k];
    dtsum += dv;
  }
  red[t] = dtsum; __syncthreads();
  // each 32-thread group == one head (256 ch / 8 ch/thr = 32 thr)
  for (int off = 16; off > 0; off >>= 1) {
    if ((t & 31) < off) red[t] += red[t + off];
    __syncthreads();
  }
  if ((t & 31) == 0) {
    float m = red[t] / (float)HDIM;
    float sp = (m > 20.f) ? m : log1pf(__expf(m));
    dt_h[(size_t)row * NH + (t >> 5)] = sp;
  }
}

// ---------------------------------------------------------------------------
// 5) per-(b,c,h) cumsum of dt*A within chunk, and tdec = exp(last)
// ---------------------------------------------------------------------------
__global__ __launch_bounds__(64) void k_cumsum(const float* __restrict__ dt_h,
                                               const float* __restrict__ A_log,
                                               float* __restrict__ lcs,
                                               float* __restrict__ tdec) {
  int bc = blockIdx.x;                 // (b*NC + c)*NH + h
  int h = bc & 7, c = (bc >> 3) & 3, b = bc >> 5;
  int l = threadIdx.x;
  __shared__ float la[LCH];
  float Ah = -__expf(A_log[h]);
  la[l] = dt_h[(size_t)(b * S_LEN + c * LCH + l) * NH + h] * Ah;
  __syncthreads();
  float sum = 0.f;
  for (int j = 0; j <= l; ++j) sum += la[j];
  lcs[(size_t)bc * LCH + l] = sum;
  if (l == LCH - 1) tdec[bc] = __expf(sum);
}

// ---------------------------------------------------------------------------
// 6) chunk kernel: per (b,c,h):  G=C*B^T (masked decay), intra=(G.M)*X,
//    statesT=(B.decay)^T*X   — all via bf16 WMMA from LDS tiles.
//    X is staged TRANSPOSED (Xt[d][l]) so B-operand fragments are two
//    contiguous ds_load_b128 instead of 16 strided ds_load_u16.
// ---------------------------------------------------------------------------
__global__ __launch_bounds__(256) void k_chunk(const float* __restrict__ proj,
                                               const bf16_t* __restrict__ xbf,
                                               const float* __restrict__ lcs,
                                               float* __restrict__ y_pre,
                                               float* __restrict__ states) {
  __shared__ __align__(16) bf16_t Cs[LCH * LCH];     // C tile   (8 KB)
  __shared__ __align__(16) bf16_t Bs2[LCH * LCH];    // B tile -> reused as G
  __shared__ __align__(16) bf16_t Bt[LCH * LCH];     // (B*decay)^T
  __shared__ __align__(16) bf16_t Xt[HDIM * LCH];    // X^T tile (32 KB)
  __shared__ float lcss[LCH];

  int bc = blockIdx.x;
  int h = bc & 7, c = (bc >> 3) & 3, b = bc >> 5;
  int t = threadIdx.x, lane = t & 31, wave = t >> 5;
  int row0 = b * S_LEN + c * LCH;

  if (t < LCH) lcss[t] = lcs[(size_t)bc * LCH + t];
  __syncthreads();
  float lcs_last = lcss[LCH - 1];

  for (int i = t; i < LCH * LCH; i += 256) {
    int l = i >> 6, n = i & 63;
    float cv = proj[(size_t)(row0 + l) * PROJ_LD + OFF_C + h * DSTATE + n];
    float bv = proj[(size_t)(row0 + l) * PROJ_LD + OFF_B + h * DSTATE + n];
    Cs[i]  = (bf16_t)cv;
    Bs2[i] = (bf16_t)bv;
    Bt[n * LCH + l] = (bf16_t)(bv * __expf(lcs_last - lcss[l]));
  }
  for (int i = t; i < LCH * HDIM; i += 256) {
    int l = i >> 8, d = i & 255;                  // coalesced global read
    Xt[d * LCH + l] = xbf[(size_t)(row0 + l) * DI + (h << 8) + d];
  }
  __syncthreads();

  // --- G = C(64x64) * B^T : 16 tiles, 2 per wave
  v8f zero = {0.f, 0.f, 0.f, 0.f, 0.f, 0.f, 0.f, 0.f};
  int tile0 = wave * 2, tile1 = wave * 2 + 1;
  int mt0 = (tile0 >> 2) << 4, nt0 = (tile0 & 3) << 4;
  int mt1 = (tile1 >> 2) << 4, nt1 = (tile1 & 3) << 4;
  v8f g0 = zero, g1 = zero;
  for (int k0 = 0; k0 < LCH; k0 += 32) {
    g0 = wmma_bf16(frag_rm(Cs, LCH, mt0, k0, lane, 63),
                   frag_rm(Bs2, LCH, nt0, k0, lane, 63), g0);
    g1 = wmma_bf16(frag_rm(Cs, LCH, mt1, k0, lane, 63),
                   frag_rm(Bs2, LCH, nt1, k0, lane, 63), g1);
  }
  __syncthreads();                 // everyone done reading Cs/Bs2
  int rlo = (lane >> 4) << 3;
#pragma unroll
  for (int r = 0; r < 8; ++r) {
    { int lr = mt0 + r + rlo, sc = nt0 + (lane & 15);
      float g = (lr >= sc) ? g0[r] * __expf(lcss[lr] - lcss[sc]) : 0.f;
      Bs2[lr * LCH + sc] = (bf16_t)g; }
    { int lr = mt1 + r + rlo, sc = nt1 + (lane & 15);
      float g = (lr >= sc) ? g1[r] * __expf(lcss[lr] - lcss[sc]) : 0.f;
      Bs2[lr * LCH + sc] = (bf16_t)g; }
  }
  __syncthreads();                 // Bs2 now holds masked G

  // --- intra = G*X  and  statesT = Bt*X   (M=64, K=64, N=256)
  int mt = (wave >> 1) << 4;       // wave's M tile
  int nb = (wave & 1) << 7;        // wave's N half (0/128)
  v16bf ga0 = frag_rm(Bs2, LCH, mt, 0,  lane, 63);
  v16bf ga1 = frag_rm(Bs2, LCH, mt, 32, lane, 63);
  v16bf ta0 = frag_rm(Bt,  LCH, mt, 0,  lane, 63);
  v16bf ta1 = frag_rm(Bt,  LCH, mt, 32, lane, 63);
  size_t sbase = (size_t)bc * (LCH * HDIM);
  for (int j = 0; j < 8; ++j) {
    int n = nb + (j << 4);
    // B-operand columns d are rows of Xt: contiguous 16B LDS loads
    v16bf x0 = frag_rm(Xt, LCH, n, 0,  lane, HDIM - 1);
    v16bf x1 = frag_rm(Xt, LCH, n, 32, lane, HDIM - 1);
    v8f ai = zero, as = zero;
    ai = wmma_bf16(ga0, x0, ai);
    ai = wmma_bf16(ga1, x1, ai);
    as = wmma_bf16(ta0, x0, as);
    as = wmma_bf16(ta1, x1, as);
    int dcol = n + (lane & 15);
#pragma unroll
    for (int r = 0; r < 8; ++r) {
      int mrow = mt + r + rlo;     // time l for intra, state n for statesT
      y_pre[(size_t)(row0 + mrow) * DI + (h << 8) + dcol] = ai[r];
      states[sbase + (size_t)mrow * HDIM + dcol] = as[r];
    }
  }
}

// ---------------------------------------------------------------------------
// 7) inter-chunk recurrence per (b,h): h0[c] = S ; S = tdec[c]*S + states[c]
//    state (64x256) kept in registers: 64 floats/thread * 256 threads
// ---------------------------------------------------------------------------
__global__ __launch_bounds__(256) void k_scan(const float* __restrict__ states,
                                              const float* __restrict__ tdec,
                                              bf16_t* __restrict__ h0) {
  int bh = blockIdx.x;             // b*NH + h
  int b = bh >> 3, h = bh & 7;
  float st[64];
#pragma unroll
  for (int i = 0; i < 64; ++i) st[i] = 0.f;
  size_t tb = (size_t)threadIdx.x * 64;
  for (int c = 0; c < NC; ++c) {
    size_t base = ((size_t)((b * NC + c) * NH + h)) * (LCH * HDIM) + tb;
#pragma unroll
    for (int i = 0; i < 64; ++i) h0[base + i] = (bf16_t)st[i];
    float td = tdec[(b * NC + c) * NH + h];
#pragma unroll
    for (int i = 0; i < 64; ++i) st[i] = td * st[i] + states[base + i];
  }
}

// ---------------------------------------------------------------------------
// 8) inter = exp(lcs[l]) * (C @ h0^T) accumulated into y_pre
//    h0 staged TRANSPOSED (St[d][n]) for contiguous B-operand fragments.
// ---------------------------------------------------------------------------
__global__ __launch_bounds__(256) void k_inter(const float* __restrict__ proj,
                                               const bf16_t* __restrict__ h0,
                                               const float* __restrict__ lcs,
                                               float* __restrict__ y_pre) {
  __shared__ __align__(16) bf16_t Cs[LCH * LCH];
  __shared__ __align__(16) bf16_t St[HDIM * LCH];   // St[d][n] = h0T transposed
  __shared__ float lcss[LCH];
  int bc = blockIdx.x;
  int h = bc & 7, c = (bc >> 3) & 3, b = bc >> 5;
  int t = threadIdx.x, lane = t & 31, wave = t >> 5;
  int row0 = b * S_LEN + c * LCH;

  if (t < LCH) lcss[t] = lcs[(size_t)bc * LCH + t];
  for (int i = t; i < LCH * LCH; i += 256) {
    int l = i >> 6, n = i & 63;
    Cs[i] = (bf16_t)proj[(size_t)(row0 + l) * PROJ_LD + OFF_C + h * DSTATE + n];
  }
  size_t hb = (size_t)bc * (LCH * HDIM);
  for (int i = t; i < LCH * HDIM; i += 256) {
    int n = i >> 8, d = i & 255;                  // h0 stored [n][d]
    St[d * LCH + n] = h0[hb + i];
  }
  __syncthreads();

  v8f zero = {0.f, 0.f, 0.f, 0.f, 0.f, 0.f, 0.f, 0.f};
  int mt = (wave >> 1) << 4;
  int nb = (wave & 1) << 7;
  v16bf a0 = frag_rm(Cs, LCH, mt, 0,  lane, 63);
  v16bf a1 = frag_rm(Cs, LCH, mt, 32, lane, 63);
  int rlo = (lane >> 4) << 3;
  for (int j = 0; j < 8; ++j) {
    int n = nb + (j << 4);
    v8f acc = zero;
    acc = wmma_bf16(a0, frag_rm(St, LCH, n, 0,  lane, HDIM - 1), acc);
    acc = wmma_bf16(a1, frag_rm(St, LCH, n, 32, lane, HDIM - 1), acc);
    int dcol = n + (lane & 15);
#pragma unroll
    for (int r = 0; r < 8; ++r) {
      int l = mt + r + rlo;
      size_t idx = (size_t)(row0 + l) * DI + (h << 8) + dcol;
      y_pre[idx] += __expf(lcss[l]) * acc[r];
    }
  }
}

// ---------------------------------------------------------------------------
// 9) y = (y_pre + x*D) * silu(z), RMSNorm(g_out) -> bf16
// ---------------------------------------------------------------------------
__global__ __launch_bounds__(256) void k_gate(const float* __restrict__ y_pre,
                                              const bf16_t* __restrict__ xbf,
                                              const float* __restrict__ proj,
                                              const float* __restrict__ Dp,
                                              const float* __restrict__ g_out,
                                              bf16_t* __restrict__ ybf) {
  int row = blockIdx.x, t = threadIdx.x;
  __shared__ float yv[DI];
  __shared__ float red[256];
  float s = 0.f;
  for (int i = t; i < DI; i += 256) {
    float y = y_pre[(size_t)row * DI + i] +
              (float)xbf[(size_t)row * DI + i] * Dp[i >> 8];
    float z = proj[(size_t)row * PROJ_LD + i];   // z block: cols [0,2048)
    y *= silu_f(z);
    yv[i] = y;
    s += y * y;
  }
  red[t] = s; __syncthreads();
  for (int off = 128; off > 0; off >>= 1) {
    if (t < off) red[t] += red[t + off];
    __syncthreads();
  }
  float rs = rsqrtf(red[0] / (float)DI + EPS_F);
  for (int i = t; i < DI; i += 256)
    ybf[(size_t)row * DI + i] = (bf16_t)(yv[i] * rs * g_out[i]);
}

// ---------------------------------------------------------------------------
extern "C" void kernel_launch(void* const* d_in, const int* in_sizes, int n_in,
                              void* d_out, int out_size, void* d_ws,
                              size_t ws_size, hipStream_t stream) {
  (void)in_sizes; (void)n_in; (void)out_size; (void)ws_size;
  const float* hidden = (const float*)d_in[0];
  const float* W_in   = (const float*)d_in[1];
  const float* conv_w = (const float*)d_in[2];
  const float* conv_b = (const float*)d_in[3];
  const float* W_dt   = (const float*)d_in[4];
  const float* b_dt   = (const float*)d_in[5];
  const float* A_log  = (const float*)d_in[6];
  const float* Dp     = (const float*)d_in[7];
  const float* W_out  = (const float*)d_in[8];
  const float* g_in   = (const float*)d_in[9];
  const float* g_out  = (const float*)d_in[10];
  float* out = (float*)d_out;

  char* ws = (char*)d_ws;
  size_t off = 0;
  auto take = [&](size_t bytes) -> char* {
    char* p = ws + off;
    off += (bytes + 255) & ~(size_t)255;
    return p;
  };
  const size_t ROWS = (size_t)B_SZ * S_LEN;                 // 8192
  bf16_t* h_bf   = (bf16_t*)take(ROWS * DM * 2);
  bf16_t* Wi_bf  = (bf16_t*)take((size_t)PROJ_N * DM * 2);
  bf16_t* Wo_bf  = (bf16_t*)take((size_t)DM * DI * 2);
  float*  proj   = (float*) take(ROWS * PROJ_LD * 4);
  bf16_t* xbf    = (bf16_t*)take(ROWS * DI * 2);
  float*  dt_h   = (float*) take(ROWS * NH * 4);
  float*  lcsb   = (float*) take((size_t)B_SZ * NC * NH * LCH * 4);
  float*  tdecb  = (float*) take((size_t)B_SZ * NC * NH * 4);
  float*  y_pre  = (float*) take(ROWS * DI * 4);
  float*  statesb= (float*) take((size_t)B_SZ * NC * NH * LCH * HDIM * 4);
  bf16_t* h0b    = (bf16_t*)take((size_t)B_SZ * NC * NH * LCH * HDIM * 2);
  bf16_t* y_bf   = (bf16_t*)take(ROWS * DI * 2);

  k_cast_bf16<<<1024, 256, 0, stream>>>(W_in, Wi_bf, (size_t)PROJ_N * DM);
  k_cast_bf16<<<1024, 256, 0, stream>>>(W_out, Wo_bf, (size_t)DM * DI);
  k_rms_in<<<(int)ROWS, 256, 0, stream>>>(hidden, g_in, h_bf);

  dim3 g1((PROJ_N + 127) / 128, (int)(ROWS / 128));
  k_gemm<<<g1, 256, 0, stream>>>(h_bf, Wi_bf, nullptr, proj,
                                 (int)ROWS, PROJ_N, DM, PROJ_LD);

  k_conv_dt<<<(int)ROWS, 256, 0, stream>>>(proj, conv_w, conv_b, W_dt, b_dt,
                                           xbf, dt_h);
  k_cumsum<<<B_SZ * NC * NH, 64, 0, stream>>>(dt_h, A_log, lcsb, tdecb);
  k_chunk<<<B_SZ * NC * NH, 256, 0, stream>>>(proj, xbf, lcsb, y_pre, statesb);
  k_scan<<<B_SZ * NH, 256, 0, stream>>>(statesb, tdecb, h0b);
  k_inter<<<B_SZ * NC * NH, 256, 0, stream>>>(proj, h0b, lcsb, y_pre);
  k_gate<<<(int)ROWS, 256, 0, stream>>>(y_pre, xbf, proj, Dp, g_out, y_bf);

  dim3 g2(DM / 128, (int)(ROWS / 128));
  k_gemm<<<g2, 256, 0, stream>>>(y_bf, Wo_bf, hidden, out,
                                 (int)ROWS, DM, DI, DM);
}
